// Mamba_8572754722871
// MI455X (gfx1250) — compile-verified, matrix operand order checked by reading, not run on hardware
//
#include <hip/hip_runtime.h>
#include <math.h>

// ---------------- types for CDNA5 WMMA ----------------
typedef __attribute__((ext_vector_type(16))) __bf16       v16bf;
typedef __attribute__((ext_vector_type(8)))  float        v8f;
typedef __attribute__((ext_vector_type(8)))  unsigned int v8u;
typedef __attribute__((ext_vector_type(4)))  int          v4i;

#if __has_builtin(__builtin_amdgcn_global_load_async_to_lds_b128)
#define HAS_ASYNC 1
#else
#define HAS_ASYNC 0
#endif

// fp32 -> packed bf16 pair (hardware cvt if available, manual RNE otherwise)
static __device__ __forceinline__ unsigned int pack2(float a, float b) {
#if __has_builtin(__builtin_amdgcn_cvt_pk_bf16_f32)
  typedef __attribute__((ext_vector_type(2))) __bf16 v2bf;
  v2bf p = __builtin_amdgcn_cvt_pk_bf16_f32(a, b);
  return __builtin_bit_cast(unsigned int, p);
#else
  unsigned int ua = __float_as_uint(a);
  ua += 0x7FFFu + ((ua >> 16) & 1u);
  unsigned int ub = __float_as_uint(b);
  ub += 0x7FFFu + ((ub >> 16) & 1u);
  return (ua >> 16) | (ub & 0xFFFF0000u);
#endif
}
static __device__ __forceinline__ unsigned short f2bf(float f) {
  return (unsigned short)(pack2(f, 0.f) & 0xFFFFu);
}

template <int N>
static __device__ __forceinline__ void wait_async() {
#if __has_builtin(__builtin_amdgcn_s_wait_asynccnt)
  __builtin_amdgcn_s_wait_asynccnt(N);
#else
  asm volatile("s_wait_asynccnt %0" :: "n"(N) : "memory");
#endif
}

// 16-byte global->LDS copy: async DMA on CDNA5, plain copy otherwise
static __device__ __forceinline__ void copy16(const unsigned short* g,
                                              unsigned int* l) {
#if HAS_ASYNC
  __builtin_amdgcn_global_load_async_to_lds_b128(
      (__attribute__((address_space(1))) v4i*)g,
      (__attribute__((address_space(3))) v4i*)l, 0, 0);
#else
  *reinterpret_cast<uint4*>(l) = *reinterpret_cast<const uint4*>(g);
#endif
}
static __device__ __forceinline__ void copy16_sync(const unsigned short* g,
                                                   unsigned int* l) {
  *reinterpret_cast<uint4*>(l) = *reinterpret_cast<const uint4*>(g);
}
static __device__ __forceinline__ void zero16(unsigned int* l) {
  l[0] = 0u; l[1] = 0u; l[2] = 0u; l[3] = 0u;
}

// ---------------- problem constants (from reference) ----------------
#define BB   2
#define LL   1024
#define DM   768
#define DI   1536
#define DST  16
#define DTR  48
#define MTOT (BB * LL)        // 2048
#define NIN  (2 * DI)         // 3072
#define NDBL (DTR + 2 * DST)  // 80
#define NDBLP 128             // padded cols of W_x^T (zero-filled)

// ---------------- tiled bf16 WMMA GEMM ----------------
// C[M,N] = A[M,K] @ B[K,N].  A: bf16 row-major [M x K] (lda).
// BT: bf16 row-major B-transpose [Npad x K] (ldb) -> staging is a pure copy.
// Block tile 64x128, 8 waves, each wave 2x2 of 16x16 WMMA tiles.
// Double-buffered LDS; async global->LDS pipeline (3 async ops/wave in flight).
// M divisible by 64 at all call sites.  EPI: 0 none, 1 softplus(acc+bias[col]).
// NG: N store guard.  KG: K tail (chunk-granular, requires K % 8 == 0).
// OUTBF: also store bf16 copy of C (for downstream GEMM consumption).
#define TM 64
#define TN 128
#define KC 32
#define LDT 20   // dwords per LDS row (16 data + 4 pad; 80B: 16B-aligned, no conflicts)

template <int EPI, bool NG, bool KG, bool OUTBF>
__global__ __launch_bounds__(256)
void wmma_gemm(const unsigned short* __restrict__ A,
               const unsigned short* __restrict__ BT,
               float* __restrict__ C,
               unsigned short* __restrict__ Cbf,
               const float* __restrict__ bias,
               int M, int N, int K, int lda, int ldb, int ldc)
{
  constexpr bool ASYNCP = (HAS_ASYNC != 0) && !KG;

  __shared__ unsigned int As[2][TM][LDT];   // [buf][row][kpair]
  __shared__ unsigned int Bs[2][TN][LDT];   // [buf][col][kpair]

  const int tid   = threadIdx.x;
  const int lane  = tid & 31;
  const int wave  = tid >> 5;
  const int waveM = wave >> 2;    // 0..1
  const int waveN = wave & 3;     // 0..3
  const int half  = lane >> 4;
  const int lo    = lane & 15;

  const int rowBase = blockIdx.y * TM;
  const int colBase = blockIdx.x * TN;

  // staging assignment: A = 1x16B per thread, B = 2x16B per thread
  const int ar = tid >> 2;              // 0..63
  const int ac = (tid & 3) * 8;         // element offset
  const int br = tid >> 1;              // 0..127
  const int bc = (tid & 1) * 16;        // element offset

  const unsigned short* aRow = A  + (size_t)(rowBase + ar) * lda;
  const unsigned short* bRow = BT + (size_t)(colBase + br) * ldb;

  auto stage = [&](int k0, int buf) {
    if (KG) {
      if (k0 + ac < K) copy16_sync(aRow + k0 + ac, &As[buf][ar][ac >> 1]);
      else             zero16(&As[buf][ar][ac >> 1]);
      #pragma unroll
      for (int c = 0; c < 2; ++c) {
        const int off = bc + c * 8;
        if (k0 + off < K) copy16_sync(bRow + k0 + off, &Bs[buf][br][off >> 1]);
        else              zero16(&Bs[buf][br][off >> 1]);
      }
    } else {
      copy16(aRow + k0 + ac,     &As[buf][ar][ac >> 1]);
      copy16(bRow + k0 + bc,     &Bs[buf][br][bc >> 1]);
      copy16(bRow + k0 + bc + 8, &Bs[buf][br][(bc >> 1) + 4]);
    }
  };

  v8f acc[2][2];
  #pragma unroll
  for (int mi = 0; mi < 2; ++mi)
    #pragma unroll
    for (int ni = 0; ni < 2; ++ni)
      #pragma unroll
      for (int v = 0; v < 8; ++v)
        acc[mi][ni][v] = 0.f;

  stage(0, 0);
  int buf = 0;
  for (int k0 = 0; k0 < K; k0 += KC) {
    const bool more = (k0 + KC) < K;
    if (more) stage(k0 + KC, buf ^ 1);
    if (ASYNCP) {
      if (more) wait_async<3>();   // allow next tile's 3 async ops in flight
      else      wait_async<0>();
    }
    __syncthreads();

    // ---- fragments: two ds_load_b128 each (conflict-free, 16B-aligned)
    v16bf afrag[2], bfrag[2];
    #pragma unroll
    for (int mi = 0; mi < 2; ++mi) {
      const uint4* arow = reinterpret_cast<const uint4*>(
          &As[buf][waveM * 32 + mi * 16 + lo][0]);
      const uint4 x0 = arow[half];       // K = koff..koff+7
      const uint4 x1 = arow[2 + half];   // K = 16+koff..16+koff+7
      v8u r;
      r[0] = x0.x; r[1] = x0.y; r[2] = x0.z; r[3] = x0.w;
      r[4] = x1.x; r[5] = x1.y; r[6] = x1.z; r[7] = x1.w;
      afrag[mi] = __builtin_bit_cast(v16bf, r);
    }
    #pragma unroll
    for (int ni = 0; ni < 2; ++ni) {
      const uint4* brow = reinterpret_cast<const uint4*>(
          &Bs[buf][waveN * 32 + ni * 16 + lo][0]);
      const uint4 x0 = brow[half];
      const uint4 x1 = brow[2 + half];
      v8u r;
      r[0] = x0.x; r[1] = x0.y; r[2] = x0.z; r[3] = x0.w;
      r[4] = x1.x; r[5] = x1.y; r[6] = x1.z; r[7] = x1.w;
      bfrag[ni] = __builtin_bit_cast(v16bf, r);
    }

    // ---- 4 WMMAs per K-chunk (A/B fragment reuse)
    #pragma unroll
    for (int mi = 0; mi < 2; ++mi)
      #pragma unroll
      for (int ni = 0; ni < 2; ++ni)
        acc[mi][ni] = __builtin_amdgcn_wmma_f32_16x16x32_bf16(
            false, afrag[mi], false, bfrag[ni], (short)0, acc[mi][ni],
            false, false);
    __syncthreads();
    buf ^= 1;
  }

  // ---- epilogue + store (C/D layout: vgpr v -> M = half*8+v, N = lo)
  #pragma unroll
  for (int mi = 0; mi < 2; ++mi)
    #pragma unroll
    for (int ni = 0; ni < 2; ++ni) {
      const int col = colBase + waveN * 32 + ni * 16 + lo;
      if (NG && col >= N) continue;
      const size_t rbase = (size_t)(rowBase + waveM * 32 + mi * 16 + half * 8);
      #pragma unroll
      for (int v = 0; v < 8; ++v) {
        float x = acc[mi][ni][v];
        if (EPI == 1) {
          x += bias[col];
          x = (x > 20.f) ? x : log1pf(expf(x));   // softplus
        }
        C[(rbase + v) * ldc + col] = x;
        if (OUTBF) Cbf[(rbase + v) * ldc + col] = f2bf(x);
      }
    }
}

// ---------------- one-time prep: fp32 -> bf16 copies ----------------
__global__ __launch_bounds__(256)
void cvt_bf16_kernel(const float* __restrict__ src,
                     unsigned short* __restrict__ dst, int n) {
  const int i = blockIdx.x * blockDim.x + threadIdx.x;
  if (i < n) dst[i] = f2bf(src[i]);
}

// W [K x N] fp32 -> WT [Npad x K] bf16 (rows n >= N zero-filled)
__global__ __launch_bounds__(256)
void transpose_bf16_kernel(const float* __restrict__ W,
                           unsigned short* __restrict__ WT,
                           int K, int N, int Npad) {
  const int idx = blockIdx.x * blockDim.x + threadIdx.x;
  if (idx >= Npad * K) return;
  const int n = idx / K;
  const int k = idx % K;
  WT[idx] = (n < N) ? f2bf(W[(size_t)k * N + n]) : (unsigned short)0;
}

// ---------------- depthwise causal conv (k=4) + bias + SiLU ----------------
__global__ __launch_bounds__(256)
void conv_silu_kernel(const float* __restrict__ xr,   // [MTOT, NIN], xi = cols [0,DI)
                      const float* __restrict__ w,    // [DI, 4]
                      const float* __restrict__ cb,   // [DI]
                      float* __restrict__ u,          // [MTOT, DI] fp32 (for scan)
                      unsigned short* __restrict__ ubf) // bf16 (for GEMM2 A)
{
  const int idx = blockIdx.x * blockDim.x + threadIdx.x;
  if (idx >= MTOT * DI) return;
  const int d  = idx % DI;
  const int bl = idx / DI;
  const int t  = bl % LL;

  float s = cb[d];
  #pragma unroll
  for (int j = 0; j < 4; ++j) {
    const int tt = t - 3 + j;
    if (tt >= 0)
      s += w[d * 4 + j] * xr[(size_t)(bl - 3 + j) * NIN + d];
  }
  const float val = s / (1.f + expf(-s));   // silu
  u[(size_t)bl * DI + d]   = val;
  ubf[(size_t)bl * DI + d] = f2bf(val);
}

// ---------------- fused selective scan ----------------
// One half-wave (16 lanes = d_state) owns one (batch, channel) recurrence.
// deltaA / deltaB_u computed on the fly (never materialized: saves ~800 MB of
// HBM traffic vs. the reference dataflow). y emitted directly as bf16 for GEMM4.
__global__ __launch_bounds__(256)
void scan_kernel(const float* __restrict__ delta,   // [MTOT, DI] (post softplus)
                 const float* __restrict__ xdbl,    // [MTOT, NDBL]
                 const float* __restrict__ xr,      // [MTOT, NIN] (res = cols [DI,NIN))
                 const float* __restrict__ A_log,   // [DI, DST]
                 const float* __restrict__ Dp,      // [DI]
                 const float* __restrict__ u,       // [MTOT, DI]
                 unsigned short* __restrict__ ybf)  // [MTOT, DI] bf16 out
{
  const int lane = threadIdx.x & 31;
  const int half = lane >> 4;
  const int n    = lane & 15;
  const int w    = blockIdx.x * (blockDim.x >> 5) + (threadIdx.x >> 5);
  const int batch = w / (DI / 2);
  const int dpair = w % (DI / 2);
  const int d     = dpair * 2 + half;

  const float Adn = -expf(A_log[d * DST + n]);
  const float Dd  = Dp[d];

  float h = 0.f;
  for (int t = 0; t < LL; ++t) {
    const size_t row = (size_t)batch * LL + t;
    const float dt = delta[row * DI + d];
    const float ut = u[row * DI + d];
    const float Bt = xdbl[row * NDBL + DTR + n];
    const float Ct = xdbl[row * NDBL + DTR + DST + n];

    h = expf(dt * Adn) * h + dt * Bt * ut;

    float c = h * Ct;                       // reduce over n within 16 lanes
    c += __shfl_xor(c, 1);
    c += __shfl_xor(c, 2);
    c += __shfl_xor(c, 4);
    c += __shfl_xor(c, 8);

    if (n == 0) {
      float y = c + ut * Dd;
      const float r = xr[row * NIN + DI + d];
      y *= r / (1.f + expf(-r));            // * silu(res)
      ybf[row * DI + d] = f2bf(y);
    }
  }
}

// ---------------- launcher ----------------
extern "C" void kernel_launch(void* const* d_in, const int* in_sizes, int n_in,
                              void* d_out, int out_size, void* d_ws, size_t ws_size,
                              hipStream_t stream) {
  (void)in_sizes; (void)n_in; (void)out_size; (void)ws_size;
  const float* x      = (const float*)d_in[0];
  const float* W_in   = (const float*)d_in[1];
  const float* conv_w = (const float*)d_in[2];
  const float* conv_b = (const float*)d_in[3];
  const float* W_x    = (const float*)d_in[4];
  const float* W_dt   = (const float*)d_in[5];
  const float* b_dt   = (const float*)d_in[6];
  const float* A_log  = (const float*)d_in[7];
  const float* Dp     = (const float*)d_in[8];
  const float* W_out  = (const float*)d_in[9];
  float* out = (float*)d_out;

  char* p = (char*)d_ws;
  auto alloc = [&](size_t bytes) {
    char* r = p;
    p += (bytes + 255) & ~(size_t)255;
    return r;
  };
  float*          xr     = (float*)         alloc((size_t)MTOT * NIN * 4);
  float*          u      = (float*)         alloc((size_t)MTOT * DI * 4);
  float*          xdbl   = (float*)         alloc((size_t)MTOT * NDBL * 4);
  float*          delta  = (float*)         alloc((size_t)MTOT * DI * 4);
  unsigned short* xbf    = (unsigned short*)alloc((size_t)MTOT * DM * 2);
  unsigned short* ubf    = (unsigned short*)alloc((size_t)MTOT * DI * 2);
  unsigned short* xdblbf = (unsigned short*)alloc((size_t)MTOT * NDBL * 2);
  unsigned short* ybf    = (unsigned short*)alloc((size_t)MTOT * DI * 2);
  unsigned short* WinT   = (unsigned short*)alloc((size_t)NIN * DM * 2);
  unsigned short* WxT    = (unsigned short*)alloc((size_t)NDBLP * DI * 2);
  unsigned short* WdtT   = (unsigned short*)alloc((size_t)DI * DTR * 2);
  unsigned short* WoutT  = (unsigned short*)alloc((size_t)DM * DI * 2);

  dim3 blk(256);
  auto g1 = [](int n) { return dim3((n + 255) / 256); };

  // ---- one-time bf16 prep (weights transposed to B^T row-major)
  cvt_bf16_kernel<<<g1(MTOT * DM), blk, 0, stream>>>(x, xbf, MTOT * DM);
  transpose_bf16_kernel<<<g1(NIN * DM),   blk, 0, stream>>>(W_in,  WinT,  DM, NIN,  NIN);
  transpose_bf16_kernel<<<g1(NDBLP * DI), blk, 0, stream>>>(W_x,   WxT,   DI, NDBL, NDBLP);
  transpose_bf16_kernel<<<g1(DI * DTR),   blk, 0, stream>>>(W_dt,  WdtT,  DTR, DI,  DI);
  transpose_bf16_kernel<<<g1(DM * DI),    blk, 0, stream>>>(W_out, WoutT, DI,  DM,  DM);

  // 1) x_and_res = x @ W_in               (2048 x 3072, K=768)
  wmma_gemm<0, false, false, false><<<dim3(NIN / TN, MTOT / TM), blk, 0, stream>>>(
      xbf, WinT, xr, nullptr, nullptr, MTOT, NIN, DM, DM, DM, NIN);

  // 2) u = silu(depthwise_conv(xi) + conv_b)   (+ bf16 copy)
  conv_silu_kernel<<<g1(MTOT * DI), blk, 0, stream>>>(xr, conv_w, conv_b, u, ubf);

  // 3) x_dbl = u @ W_x                    (2048 x 80, K=1536; padded B^T, N store guard)
  wmma_gemm<0, true, false, true><<<dim3(1, MTOT / TM), blk, 0, stream>>>(
      ubf, WxT, xdbl, xdblbf, nullptr, MTOT, NDBL, DI, DI, DI, NDBL);

  // 4) delta = softplus(x_dbl[:, :48] @ W_dt + b_dt)  (2048 x 1536, K=48 tail)
  wmma_gemm<1, false, true, false><<<dim3(DI / TN, MTOT / TM), blk, 0, stream>>>(
      xdblbf, WdtT, delta, nullptr, b_dt, MTOT, DI, DTR, NDBL, DTR, DI);

  // 5) fused selective scan (+ *silu(res)) -> ybf
  scan_kernel<<<(BB * DI / 2) / 8, blk, 0, stream>>>(delta, xdbl, xr, A_log, Dp, u, ybf);

  // 6) out = y @ W_out                    (2048 x 768, K=1536)
  wmma_gemm<0, false, false, false><<<dim3(DM / TN, MTOT / TM), blk, 0, stream>>>(
      ybf, WoutT, out, nullptr, nullptr, MTOT, DM, DI, DI, DI, DM);
}
